// StructPool_11914239279606
// MI455X (gfx1250) — matrix operand order repaired and checked by reading, not compiled, not run on hardware
//
#include <hip/hip_runtime.h>
#include <hip/hip_bf16.h>
#include <math.h>

typedef __attribute__((ext_vector_type(16))) _Float16 v16h;
typedef __attribute__((ext_vector_type(8)))  float    v8f;

constexpr int Bn = 8, Nn = 2048, Dn = 64, Kn = 16, ITERS = 5;
constexpr int KT = Nn / 32;    // 64 k-tiles of 32

// ---------------------------------------------------------------- row stats
__global__ __launch_bounds__(256) void k_rowstats(const float* __restrict__ X,
                                                  float* __restrict__ d2) {
  int r = blockIdx.x * 256 + threadIdx.x;                // 0..B*N-1
  const float4* xr = (const float4*)(X + (size_t)r * Dn);
  float acc = 0.f;
#pragma unroll
  for (int i = 0; i < Dn / 4; ++i) {
    float4 t = xr[i];
    acc += t.x * t.x + t.y * t.y + t.z * t.z + t.w * t.w;
  }
  d2[r] = acc;
}

__global__ __launch_bounds__(256) void k_colsum(const float* __restrict__ X,
                                                float* __restrict__ colsum) {
  __shared__ float red[4][Dn];
  int b = blockIdx.x;
  int d = threadIdx.x & 63, g = threadIdx.x >> 6;
  float acc = 0.f;
  for (int n = g; n < Nn; n += 4)
    acc += X[(size_t)b * Nn * Dn + (size_t)n * Dn + d];
  red[g][d] = acc;
  __syncthreads();
  if (threadIdx.x < Dn)
    colsum[b * Dn + threadIdx.x] = red[0][threadIdx.x] + red[1][threadIdx.x] +
                                   red[2][threadIdx.x] + red[3][threadIdx.x];
}

__global__ __launch_bounds__(256) void k_norm(const float* __restrict__ X,
                                              const float* __restrict__ colsum,
                                              const float* __restrict__ d2,
                                              float* __restrict__ invnm) {
  int r = blockIdx.x * 256 + threadIdx.x;
  int b = r >> 11;
  const float* xr = X + (size_t)r * Dn;
  const float* cs = colsum + b * Dn;
  float acc = 0.f;
#pragma unroll
  for (int i = 0; i < Dn; ++i) acc += xr[i] * cs[i];
  invnm[r] = 1.0f / (acc - d2[r]);
}

__global__ __launch_bounds__(256) void k_M16(const float* __restrict__ wf,
                                             const float* __restrict__ wc,
                                             float* __restrict__ M) {
  int j = threadIdx.x >> 4, k = threadIdx.x & 15;
  float acc = 0.f;
#pragma unroll
  for (int i = 0; i < Kn; ++i) acc += wf[j * Kn + i] * wc[i * Kn + k];
  M[threadIdx.x] = acc;
}

// ------------------------------------------------ X -> f16 B-fragment order
__global__ __launch_bounds__(256) void k_Xfrag(const float* __restrict__ X,
                                               _Float16* __restrict__ Xfrag) {
  int g = blockIdx.x * 256 + threadIdx.x;    // B*KT*4*32 threads
  int lane = g & 31;
  int t    = (g >> 5) & 3;
  int kt   = (g >> 7) & (KT - 1);
  int b    = g >> 13;
  int c  = lane & 15, kh = lane >> 4;
  const float* Xb = X + (size_t)b * Nn * Dn;
  v16h hv;
#pragma unroll
  for (int e = 0; e < 16; ++e) {
    int n = kt * 32 + kh * 8 + e + (e >= 8 ? 8 : 0);
    hv[e] = (_Float16)Xb[(size_t)n * Dn + t * 16 + c];
  }
  *(v16h*)(Xfrag + (size_t)g * 16) = hv;
}

// ------------------------------------------------ L -> f16 B-fragment order
__global__ __launch_bounds__(256) void k_Lfrag(const float* __restrict__ L,
                                               _Float16* __restrict__ Lfrag) {
  int g = blockIdx.x * 256 + threadIdx.x;    // B*KT*32 threads
  int lane = g & 31;
  int kt   = (g >> 5) & (KT - 1);
  int b    = g >> 11;
  int c  = lane & 15, kh = lane >> 4;
  const float* Lb = L + (size_t)b * Nn * Kn;
  v16h hv;
#pragma unroll
  for (int e = 0; e < 16; ++e) {
    int n = kt * 32 + kh * 8 + e + (e >= 8 ? 8 : 0);
    hv[e] = (_Float16)Lb[(size_t)n * Kn + c];
  }
  *(v16h*)(Lfrag + (size_t)g * 16) = hv;
}

// -------------------------------------------- U = tanh(pool(A@X)/rowsum), WMMA
__global__ __launch_bounds__(32) void k_gemm_AX_U(const float* __restrict__ A,
                                                  const _Float16* __restrict__ Xfrag,
                                                  float* __restrict__ U,
                                                  float* __restrict__ q) {
  __shared__ float sC[16 * Dn];
  int b  = blockIdx.x >> 7;
  int m0 = (blockIdx.x & 127) * 16;
  int lane = threadIdx.x;
  int r16 = lane & 15, kh = lane >> 4, col = lane & 15;
  const float* arow = A + (size_t)b * Nn * Nn + (size_t)(m0 + r16) * Nn + 8 * kh;
  const _Float16* Xf = Xfrag + (size_t)b * KT * 4 * 32 * 16;
  v8f acc0 = {}, acc1 = {}, acc2 = {}, acc3 = {};
  for (int kt = 0; kt < KT; ++kt) {
    int kk = kt * 32;
    float4 a0 = *(const float4*)(arow + kk);
    float4 a1 = *(const float4*)(arow + kk + 4);
    float4 a2 = *(const float4*)(arow + kk + 16);
    float4 a3 = *(const float4*)(arow + kk + 20);
    if (kt + 1 < KT) __builtin_prefetch(arow + kk + 32, 0, 1);
    v16h aF;
    aF[0] = (_Float16)a0.x; aF[1] = (_Float16)a0.y;
    aF[2] = (_Float16)a0.z; aF[3] = (_Float16)a0.w;
    aF[4] = (_Float16)a1.x; aF[5] = (_Float16)a1.y;
    aF[6] = (_Float16)a1.z; aF[7] = (_Float16)a1.w;
    aF[8] = (_Float16)a2.x; aF[9] = (_Float16)a2.y;
    aF[10] = (_Float16)a2.z; aF[11] = (_Float16)a2.w;
    aF[12] = (_Float16)a3.x; aF[13] = (_Float16)a3.y;
    aF[14] = (_Float16)a3.z; aF[15] = (_Float16)a3.w;
    const _Float16* bp = Xf + ((size_t)kt * 4 * 32 + lane) * 16;
    v16h bF0 = *(const v16h*)(bp);
    v16h bF1 = *(const v16h*)(bp + 32 * 16);
    v16h bF2 = *(const v16h*)(bp + 64 * 16);
    v16h bF3 = *(const v16h*)(bp + 96 * 16);
    acc0 = __builtin_amdgcn_wmma_f32_16x16x32_f16(false, aF, false, bF0, (short)0, acc0, false, false);
    acc1 = __builtin_amdgcn_wmma_f32_16x16x32_f16(false, aF, false, bF1, (short)0, acc1, false, false);
    acc2 = __builtin_amdgcn_wmma_f32_16x16x32_f16(false, aF, false, bF2, (short)0, acc2, false, false);
    acc3 = __builtin_amdgcn_wmma_f32_16x16x32_f16(false, aF, false, bF3, (short)0, acc3, false, false);
  }
#pragma unroll
  for (int r8 = 0; r8 < 8; ++r8) {
    int row = r8 + 8 * kh;
    sC[row * Dn +  0 + col] = acc0[r8];
    sC[row * Dn + 16 + col] = acc1[r8];
    sC[row * Dn + 32 + col] = acc2[r8];
    sC[row * Dn + 48 + col] = acc3[r8];
  }
  __syncthreads();
  if (lane < 16) {
    float p[Kn], ssum = 0.f;
#pragma unroll
    for (int k = 0; k < Kn; ++k) {
      float v = (sC[lane * Dn + 4 * k] + sC[lane * Dn + 4 * k + 1] +
                 sC[lane * Dn + 4 * k + 2] + sC[lane * Dn + 4 * k + 3]) * 0.25f;
      p[k] = v; ssum += v;
    }
    float inv = 1.0f / ssum;
    size_t base = ((size_t)b * Nn + m0 + lane) * Kn;
#pragma unroll
    for (int k = 0; k < Kn; ++k) {
      float u = tanhf(p[k] * inv);
      U[base + k] = u;
      q[base + k] = u;
    }
  }
}

// ---------------------------------- s = softmax(q); Z[b] = X^T s
__global__ __launch_bounds__(512) void k_soft_Z(const float* __restrict__ X,
                                                const float* __restrict__ q,
                                                float* __restrict__ s,
                                                float* __restrict__ Z) {
  __shared__ float sX[128 * Dn];
  __shared__ float sS[128 * Kn];
  int b = blockIdx.x, tid = threadIdx.x;
  int p0 = tid, p1 = tid + 512;
  int d0 = p0 >> 4, k0 = p0 & 15, d1 = p1 >> 4, k1 = p1 & 15;
  float a0 = 0.f, a1 = 0.f;
  const float* Xb = X + (size_t)b * Nn * Dn;
  for (int n0 = 0; n0 < Nn; n0 += 128) {
#pragma unroll
    for (int i = 0; i < 4; ++i) {
      int idx = tid + 512 * i;
      int rr = idx >> 4, cg = idx & 15;
      *(float4*)&sX[rr * Dn + cg * 4] =
          *(const float4*)(Xb + (size_t)(n0 + rr) * Dn + cg * 4);
    }
    if (tid < 128) {
      size_t base = ((size_t)b * Nn + n0 + tid) * Kn;
      float v[Kn], mx = -1e30f;
#pragma unroll
      for (int k = 0; k < Kn; ++k) { v[k] = q[base + k]; mx = fmaxf(mx, v[k]); }
      float se = 0.f;
#pragma unroll
      for (int k = 0; k < Kn; ++k) { v[k] = __expf(v[k] - mx); se += v[k]; }
      float inv = 1.0f / se;
#pragma unroll
      for (int k = 0; k < Kn; ++k) {
        float sv = v[k] * inv;
        sS[tid * Kn + k] = sv;
        s[base + k] = sv;
      }
    }
    __syncthreads();
    for (int r = 0; r < 128; ++r) {
      a0 += sX[r * Dn + d0] * sS[r * Kn + k0];
      a1 += sX[r * Dn + d1] * sS[r * Kn + k1];
    }
    __syncthreads();
  }
  Z[b * Dn * Kn + p0] = a0;
  Z[b * Dn * Kn + p1] = a1;
}

// --------- q = U - ((X·Z - d2*s)*invnm) @ (w_filter@w_compat)
__global__ __launch_bounds__(256) void k_update_q(const float* __restrict__ X,
                                                  const float* __restrict__ s,
                                                  const float* __restrict__ Z,
                                                  const float* __restrict__ U,
                                                  const float* __restrict__ d2,
                                                  const float* __restrict__ invnm,
                                                  const float* __restrict__ M,
                                                  float* __restrict__ q) {
  __shared__ float sZ[Dn * Kn];
  __shared__ float sM[Kn * Kn];
  int b = blockIdx.x >> 3;
  int tid = threadIdx.x;
#pragma unroll
  for (int i = 0; i < 4; ++i) sZ[tid + 256 * i] = Z[b * Dn * Kn + tid + 256 * i];
  sM[tid] = M[tid];
  __syncthreads();
  int r = blockIdx.x * 256 + tid;
  const float* xr = X + (size_t)r * Dn;
  float xv[Dn];
#pragma unroll
  for (int i = 0; i < Dn / 4; ++i) {
    float4 t = *(const float4*)(xr + 4 * i);
    xv[4 * i] = t.x; xv[4 * i + 1] = t.y; xv[4 * i + 2] = t.z; xv[4 * i + 3] = t.w;
  }
  float out[Kn];
#pragma unroll
  for (int k = 0; k < Kn; ++k) out[k] = 0.f;
  for (int dd = 0; dd < Dn; ++dd) {
    float x = xv[dd];
#pragma unroll
    for (int k = 0; k < Kn; ++k) out[k] += x * sZ[dd * Kn + k];
  }
  float dv = d2[r], iv = invnm[r];
  size_t base = (size_t)r * Kn;
#pragma unroll
  for (int k = 0; k < Kn; ++k) out[k] = (out[k] - dv * s[base + k]) * iv;
#pragma unroll
  for (int k = 0; k < Kn; ++k) {
    float acc = 0.f;
#pragma unroll
    for (int j = 0; j < Kn; ++j) acc += out[j] * sM[j * Kn + k];
    q[base + k] = U[base + k] - acc;
  }
}

__global__ __launch_bounds__(256) void k_final_L(const float* __restrict__ q,
                                                 float* __restrict__ L) {
  int r = blockIdx.x * 256 + threadIdx.x;
  size_t base = (size_t)r * Kn;
  float v[Kn], mx = -1e30f;
#pragma unroll
  for (int k = 0; k < Kn; ++k) { v[k] = q[base + k]; mx = fmaxf(mx, v[k]); }
  float se = 0.f;
#pragma unroll
  for (int k = 0; k < Kn; ++k) { v[k] = __expf(v[k] - mx); se += v[k]; }
  float inv = 1.0f / se;
#pragma unroll
  for (int k = 0; k < Kn; ++k) L[base + k] = v[k] * inv;
}

// --------------------------------------------------- X_out = L^T X
__global__ __launch_bounds__(512) void k_Xout(const float* __restrict__ X,
                                              const float* __restrict__ L,
                                              float* __restrict__ out) {
  __shared__ float sX[128 * Dn];
  __shared__ float sL[128 * Kn];
  int b = blockIdx.x, tid = threadIdx.x;
  int p0 = tid, p1 = tid + 512;
  int k0 = p0 >> 6, d0 = p0 & 63, k1 = p1 >> 6, d1 = p1 & 63;
  float a0 = 0.f, a1 = 0.f;
  const float* Xb = X + (size_t)b * Nn * Dn;
  for (int n0 = 0; n0 < Nn; n0 += 128) {
#pragma unroll
    for (int i = 0; i < 4; ++i) {
      int idx = tid + 512 * i;
      int rr = idx >> 4, cg = idx & 15;
      *(float4*)&sX[rr * Dn + cg * 4] =
          *(const float4*)(Xb + (size_t)(n0 + rr) * Dn + cg * 4);
    }
    {
      int rr = tid >> 2, cg = tid & 3;
      *(float4*)&sL[rr * Kn + cg * 4] =
          *(const float4*)(L + ((size_t)b * Nn + n0 + rr) * Kn + cg * 4);
    }
    __syncthreads();
    for (int r = 0; r < 128; ++r) {
      a0 += sL[r * Kn + k0] * sX[r * Dn + d0];
      a1 += sL[r * Kn + k1] * sX[r * Dn + d1];
    }
    __syncthreads();
  }
  out[b * Kn * Dn + p0] = a0;
  out[b * Kn * Dn + p1] = a1;
}

// --------------------------------------------------- AL = A @ L via WMMA
__global__ __launch_bounds__(32) void k_gemm_AL(const float* __restrict__ A,
                                                const _Float16* __restrict__ Lfrag,
                                                float* __restrict__ AL) {
  __shared__ float sC[16 * Kn];
  int b  = blockIdx.x >> 7;
  int m0 = (blockIdx.x & 127) * 16;
  int lane = threadIdx.x;
  int r16 = lane & 15, kh = lane >> 4, col = lane & 15;
  const float* arow = A + (size_t)b * Nn * Nn + (size_t)(m0 + r16) * Nn + 8 * kh;
  const _Float16* Lf = Lfrag + (size_t)b * KT * 32 * 16;
  v8f acc = {};
  for (int kt = 0; kt < KT; ++kt) {
    int kk = kt * 32;
    float4 a0 = *(const float4*)(arow + kk);
    float4 a1 = *(const float4*)(arow + kk + 4);
    float4 a2 = *(const float4*)(arow + kk + 16);
    float4 a3 = *(const float4*)(arow + kk + 20);
    if (kt + 1 < KT) __builtin_prefetch(arow + kk + 32, 0, 1);
    v16h aF;
    aF[0] = (_Float16)a0.x; aF[1] = (_Float16)a0.y;
    aF[2] = (_Float16)a0.z; aF[3] = (_Float16)a0.w;
    aF[4] = (_Float16)a1.x; aF[5] = (_Float16)a1.y;
    aF[6] = (_Float16)a1.z; aF[7] = (_Float16)a1.w;
    aF[8] = (_Float16)a2.x; aF[9] = (_Float16)a2.y;
    aF[10] = (_Float16)a2.z; aF[11] = (_Float16)a2.w;
    aF[12] = (_Float16)a3.x; aF[13] = (_Float16)a3.y;
    aF[14] = (_Float16)a3.z; aF[15] = (_Float16)a3.w;
    v16h bF = *(const v16h*)(Lf + ((size_t)kt * 32 + lane) * 16);
    acc = __builtin_amdgcn_wmma_f32_16x16x32_f16(false, aF, false, bF, (short)0, acc, false, false);
  }
#pragma unroll
  for (int r8 = 0; r8 < 8; ++r8) sC[(r8 + 8 * kh) * Kn + col] = acc[r8];
  __syncthreads();
  if (lane < 16) {
    size_t base = ((size_t)b * Nn + m0 + lane) * Kn;
#pragma unroll
    for (int k = 0; k < Kn; ++k) AL[base + k] = sC[lane * Kn + k];
  }
}

// --------------------------------------------------- A_out = L^T (A L)
__global__ __launch_bounds__(256) void k_Aout(const float* __restrict__ L,
                                              const float* __restrict__ AL,
                                              float* __restrict__ out) {
  __shared__ float sL[128 * Kn];
  __shared__ float sAL[128 * Kn];
  int b = blockIdx.x, tid = threadIdx.x;
  int k0 = tid >> 4, k2 = tid & 15;
  float a = 0.f;
  for (int n0 = 0; n0 < Nn; n0 += 128) {
#pragma unroll
    for (int i = 0; i < 2; ++i) {
      int idx = tid + 256 * i;
      int rr = idx >> 2, cg = idx & 3;
      *(float4*)&sL[rr * Kn + cg * 4] =
          *(const float4*)(L + ((size_t)b * Nn + n0 + rr) * Kn + cg * 4);
      *(float4*)&sAL[rr * Kn + cg * 4] =
          *(const float4*)(AL + ((size_t)b * Nn + n0 + rr) * Kn + cg * 4);
    }
    __syncthreads();
    for (int r = 0; r < 128; ++r)
      a += sL[r * Kn + k0] * sAL[r * Kn + k2];
    __syncthreads();
  }
  out[b * Kn * Kn + tid] = a;
}

// ---------------------------------------------------------------- launch
extern "C" void kernel_launch(void* const* d_in, const int* in_sizes, int n_in,
                              void* d_out, int out_size, void* d_ws, size_t ws_size,
                              hipStream_t stream) {
  const float* X  = (const float*)d_in[0];
  const float* A  = (const float*)d_in[1];
  const float* wf = (const float*)d_in[2];
  const float* wc = (const float*)d_in[3];

  float* ws   = (float*)d_ws;
  float* U    = ws;                       // 262144
  float* q    = U    + 262144;
  float* s    = q    + 262144;
  float* L    = s    + 262144;
  float* AL   = L    + 262144;
  float* d2   = AL   + 262144;            // 16384
  float* invn = d2   + 16384;             // 16384
  float* cs   = invn + 16384;             // 512
  float* Z    = cs   + 512;               // 8192
  float* M    = Z    + 8192;              // 256
  _Float16* Xfrag = (_Float16*)(M + 256);           // 1M halfs
  _Float16* Lfrag = (_Float16*)(M + 256 + 524288);  // 256K halfs

  float* Xout = (float*)d_out;            // 8192
  float* Aout = Xout + Bn * Kn * Dn;      // 2048

  k_rowstats<<<(Bn * Nn) / 256, 256, 0, stream>>>(X, d2);
  k_colsum<<<Bn, 256, 0, stream>>>(X, cs);
  k_norm<<<(Bn * Nn) / 256, 256, 0, stream>>>(X, cs, d2, invn);
  k_M16<<<1, 256, 0, stream>>>(wf, wc, M);
  k_Xfrag<<<(Bn * KT * 4 * 32) / 256, 256, 0, stream>>>(X, Xfrag);
  k_gemm_AX_U<<<Bn * (Nn / 16), 32, 0, stream>>>(A, Xfrag, U, q);

  for (int it = 0; it < ITERS; ++it) {
    k_soft_Z<<<Bn, 512, 0, stream>>>(X, q, s, Z);
    k_update_q<<<(Bn * Nn) / 256, 256, 0, stream>>>(X, s, Z, U, d2, invn, M, q);
  }

  k_final_L<<<(Bn * Nn) / 256, 256, 0, stream>>>(q, L);
  k_Lfrag<<<(Bn * KT * 32) / 256, 256, 0, stream>>>(L, Lfrag);
  k_Xout<<<Bn, 512, 0, stream>>>(X, L, Xout);
  k_gemm_AL<<<Bn * (Nn / 16), 32, 0, stream>>>(A, Lfrag, AL);
  k_Aout<<<Bn, 256, 0, stream>>>(L, AL, Aout);
}
